// MambaBackbone_19327352832715
// MI455X (gfx1250) — matrix operand order repaired and checked by reading, not compiled, not run on hardware
//
#include <hip/hip_runtime.h>
#include <hip/hip_bf16.h>

// ---------------------------------------------------------------------------
// Mamba-2 backbone forward for MI455X (gfx1250, wave32, WMMA).
// All matmul-shaped work uses v_wmma_f32_16x16x32_bf16 (bf16 in, f32 acc).
// GEMM tiles are fed with CDNA5 async copies (global_load_async_to_lds_b128,
// s_wait_asynccnt) in a double-buffered pipeline.
// Pipeline:
//   k0  fp32->bf16 conversion of x / W_in / W_out                (VALU)
//   k1  in_proj GEMM         [16384x512]x[2192x512]^T -> zxbcdt  (WMMA+async)
//   k2  conv4+silu           xBC channels                        (VALU)
//   k2b dt softplus                                              (VALU)
//   k3  per-chunk cumsum of dt*A  (block scan)                   (LDS)
//   k4  intra-chunk SSD: G=C.B^T, M=G*L, Y=M.X, states=Xd^T.B    (WMMA, 230KB LDS)
//   k5  inter-chunk state scan                                   (VALU)
//   k6  Y_off = C.prev^T * exp(cumsum)                           (WMMA)
//   k7a gated RMSNorm -> bf16                                    (VALU)
//   k7b out_proj GEMM        [16384x1024]x[512x1024]^T -> out    (WMMA+async)
// ---------------------------------------------------------------------------

#define D_MODELc 512
#define D_STATEc 64
#define D_CONVc 4
#define HEADDIMc 64
#define CHUNKc 256
#define D_INNERc 1024
#define NHEADSc 16
#define CONV_DIMc 1152
#define D_IN_PROJc 2192
#define NPROJ_PADc 2304 /* 18*128, padded row count for async W tiles */
#define BATCHc 4
#define SEQc 4096
#define NCc 16 /* SEQ / CHUNK */

typedef __attribute__((ext_vector_type(16))) __bf16 v16bf;
typedef __attribute__((ext_vector_type(8))) float v8f;

__device__ __forceinline__ v8f wmma_bf16(v16bf a, v16bf b, v8f c) {
  // (neg_a, A, neg_b, B, c_mod, C, reuse_a, reuse_b)
  return __builtin_amdgcn_wmma_f32_16x16x32_bf16(false, a, false, b, (short)0, c,
                                                 false, false);
}

// CDNA5 async copy: 16B global -> LDS, tracked by ASYNCcnt (bypasses VGPRs).
__device__ __forceinline__ void async_copy_b128(unsigned int lds_off,
                                                const void* gptr) {
  asm volatile("global_load_async_to_lds_b128 %0, %1, off" ::"v"(lds_off),
               "v"((unsigned long long)(uintptr_t)gptr)
               : "memory");
}
#define WAIT_ASYNC_4() asm volatile("s_wait_asynccnt 0x4" ::: "memory")
#define WAIT_ASYNC_0() asm volatile("s_wait_asynccnt 0x0" ::: "memory")

// A fragment (16x32 bf16, M x K). LDS layout: row-major [m][k], stride in elems.
// Lane l holds row (l&15); lanes 16-31 take K+8 base; VGPRs 0-3: K=kb+0..7,
// VGPRs 4-7: K=kb+16..23.
__device__ __forceinline__ v16bf lds_afrag(const __bf16* A, int stride, int row0,
                                           int k0, int lane) {
  const __bf16* p = A + (row0 + (lane & 15)) * stride + k0 + ((lane >> 4) << 3);
  v16bf a;
#pragma unroll
  for (int e = 0; e < 8; ++e) a[e] = p[e];
#pragma unroll
  for (int e = 8; e < 16; ++e) a[e] = p[e + 8];
  return a;
}

// B fragment (32x16 bf16, K x N), LDS stored row-major [k][n].
// Lane l holds column (l&15); lanes 16-31 take K+8 base; elems 0-7: K=kb+0..7, elems 8-15: K=kb+16..23.
__device__ __forceinline__ v16bf lds_bfrag_kn(const __bf16* B, int stride, int k0,
                                              int n0, int lane) {
  const __bf16* p = B + (k0 + ((lane >> 4) << 3)) * stride + n0 + (lane & 15);
  v16bf b;
#pragma unroll
  for (int e = 0; e < 16; ++e) b[e] = p[(e < 8 ? e : e + 8) * stride];
  return b;
}

// B fragment where LDS holds B transposed: [n][k] row-major (contiguous K).
__device__ __forceinline__ v16bf lds_bfrag_nk(const __bf16* Bt, int stride, int k0,
                                              int n0, int lane) {
  const __bf16* p = Bt + (n0 + (lane & 15)) * stride + k0 + ((lane >> 4) << 3);
  v16bf b;
#pragma unroll
  for (int e = 0; e < 16; ++e) b[e] = p[e < 8 ? e : e + 8];
  return b;
}

// ---------------------------------------------------------------------------
// k0: fp32 -> bf16 pack (one element per thread).
// ---------------------------------------------------------------------------
__global__ __launch_bounds__(256) void cvt_bf16_kernel(const float* __restrict__ src,
                                                       __bf16* __restrict__ dst,
                                                       size_t n) {
  size_t i = (size_t)blockIdx.x * 256 + threadIdx.x;
  if (i < n) dst[i] = (__bf16)src[i];
}

// ---------------------------------------------------------------------------
// k1 / k7b: C[M][N] = A[M][K] (bf16) * W[N][K]^T (bf16), f32 accumulate.
// 256 threads = 8 waves as 2(M) x 4(N); wave tile 64x32 = 4x2 WMMA tiles.
// Block tile 128x128, K-step 32. Tiles stream through LDS via async copies,
// double-buffered: s_wait_asynccnt 4 retires the compute buffer while the
// prefetch stage (4 async instructions/wave) stays in flight.
// LDS stride 40 elems (80B): 16B-aligned rows, conflict-free mod-64 banks.
// ---------------------------------------------------------------------------
__global__ __launch_bounds__(256) void gemm_wmma_bf16_async(
    const __bf16* __restrict__ A, const __bf16* __restrict__ W,
    float* __restrict__ C, int M, int N, int K) {
  __shared__ __align__(16) __bf16 Asm[2][128 * 40];
  __shared__ __align__(16) __bf16 Wsm[2][128 * 40];
  const int tid = threadIdx.x;
  const int lane = tid & 31;
  const int wid = tid >> 5;
  const int wm = wid >> 2;  // 0..1
  const int wn = wid & 3;   // 0..3
  const int m0 = blockIdx.x * 128;
  const int n0 = blockIdx.y * 128;

  v8f vz = {};
  v8f acc[4][2];
#pragma unroll
  for (int i = 0; i < 4; ++i)
#pragma unroll
    for (int j = 0; j < 2; ++j) acc[i][j] = vz;

  // Each thread owns 2 x 16B chunks of the A tile and 2 of the W tile.
  // chunk = tid*2+j over 512 chunks; row = chunk>>2, col8 = (chunk&3)*8.
  const int ch0 = tid * 2;
  const int rA0 = ch0 >> 2, cA0 = (ch0 & 3) << 3;
  const int rA1 = (ch0 + 1) >> 2, cA1 = ((ch0 + 1) & 3) << 3;

  const int nIter = K >> 5;
  {  // prologue: stage 0 into buffer 0
    async_copy_b128((unsigned int)(uintptr_t)&Asm[0][rA0 * 40 + cA0],
                    A + (size_t)(m0 + rA0) * K + cA0);
    async_copy_b128((unsigned int)(uintptr_t)&Wsm[0][rA0 * 40 + cA0],
                    W + (size_t)(n0 + rA0) * K + cA0);
    async_copy_b128((unsigned int)(uintptr_t)&Asm[0][rA1 * 40 + cA1],
                    A + (size_t)(m0 + rA1) * K + cA1);
    async_copy_b128((unsigned int)(uintptr_t)&Wsm[0][rA1 * 40 + cA1],
                    W + (size_t)(n0 + rA1) * K + cA1);
  }
  for (int it = 0; it < nIter; ++it) {
    const int cur = it & 1;
    if (it + 1 < nIter) {
      const int nxt = cur ^ 1;
      const int k0 = (it + 1) << 5;
      async_copy_b128((unsigned int)(uintptr_t)&Asm[nxt][rA0 * 40 + cA0],
                      A + (size_t)(m0 + rA0) * K + k0 + cA0);
      async_copy_b128((unsigned int)(uintptr_t)&Wsm[nxt][rA0 * 40 + cA0],
                      W + (size_t)(n0 + rA0) * K + k0 + cA0);
      async_copy_b128((unsigned int)(uintptr_t)&Asm[nxt][rA1 * 40 + cA1],
                      A + (size_t)(m0 + rA1) * K + k0 + cA1);
      async_copy_b128((unsigned int)(uintptr_t)&Wsm[nxt][rA1 * 40 + cA1],
                      W + (size_t)(n0 + rA1) * K + k0 + cA1);
      WAIT_ASYNC_4();  // retire current stage, keep prefetch in flight
    } else {
      WAIT_ASYNC_0();
    }
    __syncthreads();  // current-stage tiles visible to all waves
    v16bf af[4], bf[2];
#pragma unroll
    for (int mt = 0; mt < 4; ++mt)
      af[mt] = lds_afrag(Asm[cur], 40, wm * 64 + mt * 16, 0, lane);
#pragma unroll
    for (int nt = 0; nt < 2; ++nt)
      bf[nt] = lds_bfrag_nk(Wsm[cur], 40, 0, wn * 32 + nt * 16, lane);
#pragma unroll
    for (int mt = 0; mt < 4; ++mt)
#pragma unroll
      for (int nt = 0; nt < 2; ++nt)
        acc[mt][nt] = wmma_bf16(af[mt], bf[nt], acc[mt][nt]);
    __syncthreads();  // all reads done before this buffer is refilled
  }

  const int rb = (lane >> 4) << 3;
#pragma unroll
  for (int mt = 0; mt < 4; ++mt)
#pragma unroll
    for (int nt = 0; nt < 2; ++nt) {
      int col = n0 + wn * 32 + nt * 16 + (lane & 15);
      if (col < N) {
#pragma unroll
        for (int i = 0; i < 8; ++i) {
          int row = m0 + wm * 64 + mt * 16 + rb + i;
          C[(size_t)row * N + col] = acc[mt][nt][i];
        }
      }
    }
}

// ---------------------------------------------------------------------------
// k2: causal depthwise conv (4 taps) + SiLU on xBC channels.
// ---------------------------------------------------------------------------
__global__ __launch_bounds__(256) void conv_silu_kernel(
    const float* __restrict__ zxbcdt, const float* __restrict__ conv_w,
    const float* __restrict__ conv_b, float* __restrict__ xBCc) {
  int idx = blockIdx.x * 256 + threadIdx.x;  // over B*T*CONV_DIM
  int c = idx % CONV_DIMc;
  int bt = idx / CONV_DIMc;
  int t = bt % SEQc;
  int b = bt / SEQc;
  float acc = conv_b[c];
#pragma unroll
  for (int k = 0; k < D_CONVc; ++k) {
    int tt = t + k - (D_CONVc - 1);
    if (tt >= 0)
      acc += zxbcdt[(size_t)(b * SEQc + tt) * D_IN_PROJc + D_INNERc + c] *
             conv_w[c * D_CONVc + k];
  }
  xBCc[(size_t)(b * SEQc + t) * CONV_DIMc + c] = acc / (1.f + __expf(-acc));
}

// ---------------------------------------------------------------------------
// k2b: dt = softplus(dt_raw + dt_bias)
// ---------------------------------------------------------------------------
__global__ __launch_bounds__(256) void dt_softplus_kernel(
    const float* __restrict__ zxbcdt, const float* __restrict__ dt_bias,
    float* __restrict__ dt_soft) {
  int idx = blockIdx.x * 256 + threadIdx.x;  // over B*T*NHEADS
  int h = idx % NHEADSc;
  int bt = idx / NHEADSc;
  float v = zxbcdt[(size_t)bt * D_IN_PROJc + (D_IN_PROJc - NHEADSc) + h] + dt_bias[h];
  dt_soft[idx] = (v > 20.f) ? v : log1pf(__expf(v));
}

// ---------------------------------------------------------------------------
// k3: inclusive scan of dt*A over each chunk; block = one (b,h,c).
// A_cs layout: [b][h][c][l]; chunk_decay[b][h][c] = exp(cs[255]).
// ---------------------------------------------------------------------------
__global__ __launch_bounds__(256) void cumsum_kernel(
    const float* __restrict__ dt_soft, const float* __restrict__ A_log,
    float* __restrict__ A_cs, float* __restrict__ chunk_dec) {
  int bid = blockIdx.x;  // (b*NHEADS + h)*NC + c
  int c = bid % NCc;
  int h = (bid / NCc) % NHEADSc;
  int b = bid / (NCc * NHEADSc);
  int l = threadIdx.x;
  float Ah = -__expf(A_log[h]);
  float v = dt_soft[(size_t)(b * SEQc + c * CHUNKc + l) * NHEADSc + h] * Ah;
  __shared__ float buf[256];
  buf[l] = v;
  __syncthreads();
  for (int off = 1; off < 256; off <<= 1) {
    float add = (l >= off) ? buf[l - off] : 0.f;
    __syncthreads();
    buf[l] += add;
    __syncthreads();
  }
  A_cs[(size_t)bid * CHUNKc + l] = buf[l];
  if (l == 255) chunk_dec[bid] = __expf(buf[255]);
}

// ---------------------------------------------------------------------------
// k4: intra-chunk SSD, one block per (b,c,h); 512 threads = 16 waves.
// Dynamic LDS: csA[256] f32, dec[256] f32, Cbf/Bbf/Xbf [256][66] bf16,
// Mbf [256][258] bf16   (total 235,520 B; fits 320 KB WGP LDS).
// ---------------------------------------------------------------------------
__global__ __launch_bounds__(512) void ssd_chunk_kernel(
    const float* __restrict__ xBCc, const float* __restrict__ dt_soft,
    const float* __restrict__ A_cs, const float* __restrict__ Dptr,
    float* __restrict__ ybuf, float* __restrict__ states) {
  extern __shared__ char smem[];
  float* csA = (float*)smem;           // 256
  float* dec = csA + 256;              // 256
  __bf16* Cbf = (__bf16*)(dec + 256);  // [256][66]
  __bf16* Bbf = Cbf + 256 * 66;
  __bf16* Xbf = Bbf + 256 * 66;  // x*dt, [s][p]
  __bf16* Mbf = Xbf + 256 * 66;  // masked L*(C.B^T), [l][s], stride 258

  const int bid = blockIdx.x;  // ((b*NC)+c)*NHEADS + h
  const int h = bid & 15;
  const int c = (bid >> 4) & 15;
  const int b = bid >> 8;
  const int tid = threadIdx.x;
  const int lane = tid & 31;
  const int w = tid >> 5;  // 0..15

  if (tid < 256) csA[tid] = A_cs[(((size_t)(b * NHEADSc + h) * NCc + c) << 8) + tid];
  for (int i = tid; i < 256 * 64; i += 512) {
    int s = i >> 6, j = i & 63;
    int t = c * CHUNKc + s;
    const float* rowp = xBCc + (size_t)(b * SEQc + t) * CONV_DIMc;
    Cbf[s * 66 + j] = (__bf16)rowp[D_INNERc + D_STATEc + j];
    Bbf[s * 66 + j] = (__bf16)rowp[D_INNERc + j];
    float dtv = dt_soft[(size_t)(b * SEQc + t) * NHEADSc + h];
    Xbf[s * 66 + j] = (__bf16)(rowp[h * HEADDIMc + j] * dtv);
  }
  __syncthreads();
  if (tid < 256) dec[tid] = __expf(csA[255] - csA[tid]);
  __syncthreads();

  // ---- Stage 1: G = C (256x64) @ B^T (64x256); wave w owns 16 l-rows ----
  v8f vz = {};
  v8f g[16];
#pragma unroll
  for (int ts = 0; ts < 16; ++ts) g[ts] = vz;
#pragma unroll
  for (int k0 = 0; k0 < 64; k0 += 32) {
    v16bf af = lds_afrag(Cbf, 66, w * 16, k0, lane);
#pragma unroll
    for (int ts = 0; ts < 16; ++ts) {
      // B-matrix element [k=n][col=s] = Bc[s][n]: transposed-layout loader.
      v16bf bfv = lds_bfrag_nk(Bbf, 66, k0, ts * 16, lane);
      g[ts] = wmma_bf16(af, bfv, g[ts]);
    }
  }
  // Mask with L = exp(segsum), write bf16 M to LDS.
  const int rb = (lane >> 4) << 3;
  const int sc = lane & 15;
#pragma unroll
  for (int ts = 0; ts < 16; ++ts) {
#pragma unroll
    for (int i = 0; i < 8; ++i) {
      int l = w * 16 + rb + i;
      int s = ts * 16 + sc;
      float val = (s <= l) ? g[ts][i] * __expf(csA[l] - csA[s]) : 0.f;
      Mbf[l * 258 + s] = (__bf16)val;
    }
  }
  __syncthreads();

  // ---- Stage 2: Y_diag = M (256x256) @ X (256x64) ----
  v8f yacc[4];
#pragma unroll
  for (int nt = 0; nt < 4; ++nt) yacc[nt] = vz;
  for (int s0 = 0; s0 < 256; s0 += 32) {
    v16bf af = lds_afrag(Mbf, 258, w * 16, s0, lane);
#pragma unroll
    for (int nt = 0; nt < 4; ++nt) {
      v16bf bfv = lds_bfrag_kn(Xbf, 66, s0, nt * 16, lane);
      yacc[nt] = wmma_bf16(af, bfv, yacc[nt]);
    }
  }
  float Dh = Dptr[h];
#pragma unroll
  for (int nt = 0; nt < 4; ++nt) {
    int p = nt * 16 + sc;
#pragma unroll
    for (int i = 0; i < 8; ++i) {
      int l = w * 16 + rb + i;
      int t = c * CHUNKc + l;
      float xs = xBCc[(size_t)(b * SEQc + t) * CONV_DIMc + h * HEADDIMc + p];
      ybuf[(size_t)(b * SEQc + t) * D_INNERc + h * HEADDIMc + p] = yacc[nt][i] + Dh * xs;
    }
  }

  // ---- Stage 3: states[p][n] = sum_s dec[s]*X[s][p]*B[s][n]  (64x64) ----
  const int pt = w >> 2, ntile = w & 3;  // 4x4 tiles over 16 waves
  v8f st = vz;
  for (int s0 = 0; s0 < 256; s0 += 32) {
    v16bf af;
    {
      int p = pt * 16 + (lane & 15);
      int kb = s0 + ((lane >> 4) << 3);
#pragma unroll
      for (int e = 0; e < 8; ++e) {
        int s = kb + e;
        af[e] = (__bf16)((float)Xbf[s * 66 + p] * dec[s]);
      }
#pragma unroll
      for (int e = 8; e < 16; ++e) {
        int s = kb + 8 + e;
        af[e] = (__bf16)((float)Xbf[s * 66 + p] * dec[s]);
      }
    }
    v16bf bfv = lds_bfrag_kn(Bbf, 66, s0, ntile * 16, lane);
    st = wmma_bf16(af, bfv, st);
  }
  {
    int n = ntile * 16 + sc;
#pragma unroll
    for (int i = 0; i < 8; ++i) {
      int p = pt * 16 + rb + i;
      states[((((size_t)c * BATCHc + b) * NHEADSc + h) * HEADDIMc + p) * D_STATEc + n] =
          st[i];
    }
  }
}

// ---------------------------------------------------------------------------
// k5: sequential scan over 16 chunks; parallel over (b,h,p,n).
// ---------------------------------------------------------------------------
__global__ __launch_bounds__(256) void chunk_scan_kernel(
    const float* __restrict__ states, const float* __restrict__ chunk_dec,
    float* __restrict__ prev) {
  int bid = blockIdx.x;  // (b*NHEADS+h)*16 + part
  int bh = bid >> 4;
  int part = bid & 15;
  int b = bh >> 4;
  int h = bh & 15;
  int e = part * 256 + threadIdx.x;  // element within 64x64 state
  float hval = 0.f;
#pragma unroll
  for (int c = 0; c < NCc; ++c) {
    size_t idx = (((size_t)c * BATCHc + b) * NHEADSc + h) * 4096 + e;
    prev[idx] = hval;
    float d = chunk_dec[(b * NHEADSc + h) * NCc + c];
    hval = d * hval + states[idx];
  }
}

// ---------------------------------------------------------------------------
// k6: Y_off = exp(A_cs[l]) * (C (256x64) @ prev^T (64x64)); accumulate into y.
// One block per (b,c,h); 256 threads = 8 waves; wave owns 32 l-rows.
// ---------------------------------------------------------------------------
__global__ __launch_bounds__(256) void yoff_kernel(
    const float* __restrict__ xBCc, const float* __restrict__ A_cs,
    const float* __restrict__ prev, float* __restrict__ ybuf) {
  __shared__ __bf16 Cbf[256 * 66];
  __shared__ __bf16 Pbf[64 * 66];  // prev as [p][n]
  __shared__ float csE[256];
  const int bid = blockIdx.x;
  const int h = bid & 15;
  const int c = (bid >> 4) & 15;
  const int b = bid >> 8;
  const int tid = threadIdx.x;
  const int lane = tid & 31;
  const int w = tid >> 5;  // 0..7

  csE[tid] = __expf(A_cs[(((size_t)(b * NHEADSc + h) * NCc + c) << 8) + tid]);
  for (int i = tid; i < 256 * 64; i += 256) {
    int s = i >> 6, j = i & 63;
    int t = c * CHUNKc + s;
    Cbf[s * 66 + j] =
        (__bf16)xBCc[(size_t)(b * SEQc + t) * CONV_DIMc + D_INNERc + D_STATEc + j];
  }
  for (int i = tid; i < 64 * 64; i += 256) {
    int p = i >> 6, n = i & 63;
    Pbf[p * 66 + n] =
        (__bf16)prev[((((size_t)c * BATCHc + b) * NHEADSc + h) * HEADDIMc + p) *
                         D_STATEc + n];
  }
  __syncthreads();

  v8f vz = {};
  v8f acc[2][4];
#pragma unroll
  for (int mt = 0; mt < 2; ++mt)
#pragma unroll
    for (int nt = 0; nt < 4; ++nt) acc[mt][nt] = vz;
#pragma unroll
  for (int k0 = 0; k0 < 64; k0 += 32) {
    v16bf af[2];
#pragma unroll
    for (int mt = 0; mt < 2; ++mt)
      af[mt] = lds_afrag(Cbf, 66, w * 32 + mt * 16, k0, lane);
#pragma unroll
    for (int nt = 0; nt < 4; ++nt) {
      // B-matrix element [k=n][col=p] = prev[p][n]: transposed-layout loader.
      v16bf bfv = lds_bfrag_nk(Pbf, 66, k0, nt * 16, lane);
#pragma unroll
      for (int mt = 0; mt < 2; ++mt) acc[mt][nt] = wmma_bf16(af[mt], bfv, acc[mt][nt]);
    }
  }
  const int rb = (lane >> 4) << 3;
  const int sc = lane & 15;
#pragma unroll
  for (int mt = 0; mt < 2; ++mt)
#pragma unroll
    for (int nt = 0; nt < 4; ++nt) {
      int p = nt * 16 + sc;
#pragma unroll
      for (int i = 0; i < 8; ++i) {
        int l = w * 32 + mt * 16 + rb + i;
        int t = c * CHUNKc + l;
        size_t yi = (size_t)(b * SEQc + t) * D_INNERc + h * HEADDIMc + p;
        ybuf[yi] += csE[l] * acc[mt][nt][i];
      }
    }
}

// ---------------------------------------------------------------------------
// k7a: gated RMSNorm: yz = y*silu(z); out = yz*rsqrt(mean(yz^2)+1e-5)*norm_w.
// One block per token row; writes bf16 for the async out_proj GEMM.
// ---------------------------------------------------------------------------
__global__ __launch_bounds__(256) void norm_kernel(
    const float* __restrict__ ybuf, const float* __restrict__ zxbcdt,
    const float* __restrict__ norm_w, __bf16* __restrict__ yzn) {
  int row = blockIdx.x;
  int tid = threadIdx.x;
  float yz[4];
  float ss = 0.f;
#pragma unroll
  for (int k = 0; k < 4; ++k) {
    int j = tid + k * 256;
    float y = ybuf[(size_t)row * D_INNERc + j];
    float z = zxbcdt[(size_t)row * D_IN_PROJc + j];
    yz[k] = y * (z / (1.f + __expf(-z)));
    ss += yz[k] * yz[k];
  }
  __shared__ float red[256];
  red[tid] = ss;
  __syncthreads();
  for (int s = 128; s > 0; s >>= 1) {
    if (tid < s) red[tid] += red[tid + s];
    __syncthreads();
  }
  float rs = rsqrtf(red[0] * (1.f / (float)D_INNERc) + 1e-5f);
#pragma unroll
  for (int k = 0; k < 4; ++k) {
    int j = tid + k * 256;
    yzn[(size_t)row * D_INNERc + j] = (__bf16)(yz[k] * rs * norm_w[j]);
  }
}

// ---------------------------------------------------------------------------
extern "C" void kernel_launch(void* const* d_in, const int* in_sizes, int n_in,
                              void* d_out, int out_size, void* d_ws, size_t ws_size,
                              hipStream_t stream) {
  const float* x = (const float*)d_in[0];
  const float* W_in = (const float*)d_in[1];
  const float* conv_w = (const float*)d_in[2];
  const float* conv_b = (const float*)d_in[3];
  const float* dt_bias = (const float*)d_in[4];
  const float* A_log = (const float*)d_in[5];
  const float* Dp = (const float*)d_in[6];
  const float* norm_w = (const float*)d_in[7];
  const float* W_out = (const float*)d_in[8];
  float* out = (float*)d_out;

  const size_t M = (size_t)BATCHc * SEQc;  // 16384
  float* ws = (float*)d_ws;
  float* zxbcdt = ws;                    // M*2192
  float* xBCc = zxbcdt + M * D_IN_PROJc; // M*1152
  float* dt_s = xBCc + M * CONV_DIMc;    // M*16
  float* A_cs = dt_s + M * NHEADSc;      // 4*16*16*256
  float* cdec = A_cs + (size_t)BATCHc * NHEADSc * NCc * CHUNKc;  // 1024
  float* states = cdec + BATCHc * NHEADSc * NCc;                 // 16*4*16*64*64
  float* prev = states + (size_t)NCc * BATCHc * NHEADSc * 64 * 64;
  float* ybuf = prev + (size_t)NCc * BATCHc * NHEADSc * 64 * 64;  // M*1024
  // bf16 staging regions
  __bf16* xbf = (__bf16*)(ybuf + M * D_INNERc);      // M*512
  __bf16* wbf_in = xbf + M * D_MODELc;               // 2304*512 (padded rows)
  __bf16* wbf_out = wbf_in + (size_t)NPROJ_PADc * D_MODELc;  // 512*1024
  __bf16* yznbf = (__bf16*)xBCc;  // xBCc is dead after k6; reuse (bf16 M*1024)

  // 0. pack bf16 operands for the async-fed GEMMs
  cvt_bf16_kernel<<<(int)((M * D_MODELc + 255) / 256), 256, 0, stream>>>(
      x, xbf, M * D_MODELc);
  cvt_bf16_kernel<<<(int)(((size_t)D_IN_PROJc * D_MODELc + 255) / 256), 256, 0,
                    stream>>>(W_in, wbf_in, (size_t)D_IN_PROJc * D_MODELc);
  cvt_bf16_kernel<<<(int)(((size_t)D_MODELc * D_INNERc + 255) / 256), 256, 0,
                    stream>>>(W_out, wbf_out, (size_t)D_MODELc * D_INNERc);

  // 1. in_proj GEMM: [16384,512] x [2192,512]^T  (W padded to 2304 rows)
  gemm_wmma_bf16_async<<<dim3(128, 18), 256, 0, stream>>>(xbf, wbf_in, zxbcdt, (int)M,
                                                          D_IN_PROJc, D_MODELc);
  // 2. conv + silu, dt softplus
  conv_silu_kernel<<<(int)(M * CONV_DIMc / 256), 256, 0, stream>>>(zxbcdt, conv_w,
                                                                   conv_b, xBCc);
  dt_softplus_kernel<<<(int)(M * NHEADSc / 256), 256, 0, stream>>>(zxbcdt, dt_bias,
                                                                   dt_s);
  // 3. per-chunk cumsum of dt*A
  cumsum_kernel<<<BATCHc * NHEADSc * NCc, 256, 0, stream>>>(dt_s, A_log, A_cs, cdec);
  // 4. intra-chunk SSD (WMMA heavy, 230 KB dynamic LDS)
  size_t smem4 =
      2 * 256 * sizeof(float) + (size_t)(3 * 256 * 66 + 256 * 258) * sizeof(__bf16);
  hipFuncSetAttribute((const void*)ssd_chunk_kernel,
                      hipFuncAttributeMaxDynamicSharedMemorySize, (int)smem4);
  ssd_chunk_kernel<<<BATCHc * NCc * NHEADSc, 512, smem4, stream>>>(xBCc, dt_s, A_cs,
                                                                   Dp, ybuf, states);
  // 5. inter-chunk scan
  chunk_scan_kernel<<<BATCHc * NHEADSc * 16, 256, 0, stream>>>(states, cdec, prev);
  // 6. Y_off accumulate
  yoff_kernel<<<BATCHc * NCc * NHEADSc, 256, 0, stream>>>(xBCc, A_cs, prev, ybuf);
  // 7. gated RMSNorm (bf16 out) + out_proj GEMM
  norm_kernel<<<(int)M, 256, 0, stream>>>(ybuf, zxbcdt, norm_w, yznbf);
  gemm_wmma_bf16_async<<<dim3(128, 4), 256, 0, stream>>>(yznbf, wbf_out, out, (int)M,
                                                         D_MODELc, D_INNERc);
}
